// VQCFeatureProcessor_82042465288192
// MI455X (gfx1250) — compile-verified, matrix operand order checked by reading, not run on hardware
//
#include <hip/hip_runtime.h>
#include <hip/hip_bf16.h>

#define NQ 8
#define NLAYERS 3
#define DDIM 256
#define HDIM 128

typedef __attribute__((ext_vector_type(16))) __bf16 v16bf;
typedef __attribute__((ext_vector_type(8)))  float  v8f;

// ---------------- quantum gate helpers ----------------
// State layout: amplitude index i = lane*8 + k  (k = 0..7).
// Qubit q in {0,1,2}  <-> bit q of k        (intra-lane register pairs)
// Qubit q in {3..7}   <-> bit (q-3) of lane (cross-lane via shfl_xor)

__device__ __forceinline__ void ry_gate(float aR[8], float aI[8], int q,
                                        float c, float s, int lane) {
    if (q < 3) {
        const int m = 1 << q;
        #pragma unroll
        for (int k = 0; k < 8; ++k) {
            if ((k & m) == 0) {
                const int k1 = k | m;
                float r0 = aR[k], i0 = aI[k], r1 = aR[k1], i1 = aI[k1];
                aR[k]  = c * r0 - s * r1;   aI[k]  = c * i0 - s * i1;
                aR[k1] = s * r0 + c * r1;   aI[k1] = s * i0 + c * i1;
            }
        }
    } else {
        const int lm = 1 << (q - 3);
        const bool hi = (lane & lm) != 0;
        #pragma unroll
        for (int k = 0; k < 8; ++k) {
            float pr = __shfl_xor(aR[k], lm, 32);
            float pi = __shfl_xor(aI[k], lm, 32);
            // bit==0 lane holds a0: a0' = c*a0 - s*a1 ; bit==1 lane holds a1: a1' = s*a0 + c*a1
            aR[k] = hi ? (s * pr + c * aR[k]) : (c * aR[k] - s * pr);
            aI[k] = hi ? (s * pi + c * aI[k]) : (c * aI[k] - s * pi);
        }
    }
}

__device__ __forceinline__ void rz_gate(float aR[8], float aI[8], int q,
                                        float c, float s, int lane) {
    #pragma unroll
    for (int k = 0; k < 8; ++k) {
        const int bit = (q < 3) ? ((k >> q) & 1) : ((lane >> (q - 3)) & 1);
        const float t = bit ? s : -s;          // phase = c + i*t  (RZ: diag(e^{-ip/2}, e^{+ip/2}))
        float r = aR[k], im = aI[k];
        aR[k] = r * c - im * t;
        aI[k] = im * c + r * t;
    }
}

__device__ __forceinline__ void cnot_gate(float aR[8], float aI[8], int cq, int tq, int lane) {
    if (tq < 3) {
        const int tm = 1 << tq;
        if (cq < 3) {
            const int cm = 1 << cq;
            #pragma unroll
            for (int k = 0; k < 8; ++k) {
                if ((k & cm) && !(k & tm)) {
                    const int k1 = k | tm;
                    float r = aR[k]; aR[k] = aR[k1]; aR[k1] = r;
                    float i = aI[k]; aI[k] = aI[k1]; aI[k1] = i;
                }
            }
        } else {
            const int clm = 1 << (cq - 3);
            const bool ctrl = (lane & clm) != 0;
            #pragma unroll
            for (int k = 0; k < 8; ++k) {
                if (!(k & tm)) {
                    const int k1 = k | tm;
                    float r0 = aR[k], i0 = aI[k], r1 = aR[k1], i1 = aI[k1];
                    aR[k]  = ctrl ? r1 : r0;  aI[k]  = ctrl ? i1 : i0;
                    aR[k1] = ctrl ? r0 : r1;  aI[k1] = ctrl ? i0 : i1;
                }
            }
        }
    } else {
        const int tlm = 1 << (tq - 3);
        #pragma unroll
        for (int k = 0; k < 8; ++k) {
            float pr = __shfl_xor(aR[k], tlm, 32);
            float pi = __shfl_xor(aI[k], tlm, 32);
            const bool ctrl = (cq < 3) ? (((k >> cq) & 1) != 0)
                                       : (((lane >> (cq - 3)) & 1) != 0);
            if (ctrl) { aR[k] = pr; aI[k] = pi; }
        }
    }
}

// ---------------- fused kernel: 16 rows per block, 512 threads (16 waves) ----------------
__global__ __launch_bounds__(512)
void vqc_fused_kernel(const float* __restrict__ x,
                      const float* __restrict__ theta,
                      const float* __restrict__ phi,
                      const float* __restrict__ W1,
                      const float* __restrict__ b1,
                      const float* __restrict__ W2,
                      const float* __restrict__ b2,
                      float* __restrict__ out)
{
    __shared__ float  qLDS[16 * NQ];
    __shared__ __bf16 hhi[16 * HDIM];
    __shared__ __bf16 hlo[16 * HDIM];

    const int tid  = threadIdx.x;
    const int wave = tid >> 5;          // 0..15: row within block (circuit) / N-tile (wmma)
    const int lane = tid & 31;
    const long row = (long)blockIdx.x * 16 + wave;

    // ---- load this row's 8-float slice; row min/max via wave butterfly ----
    const float4* xr4 = reinterpret_cast<const float4*>(x + row * DDIM + lane * 8);
    const float4 a0 = xr4[0];
    const float4 a1 = xr4[1];

    float mn = fminf(fminf(fminf(a0.x, a0.y), fminf(a0.z, a0.w)),
                     fminf(fminf(a1.x, a1.y), fminf(a1.z, a1.w)));
    float mx = fmaxf(fmaxf(fmaxf(a0.x, a0.y), fmaxf(a0.z, a0.w)),
                     fmaxf(fmaxf(a1.x, a1.y), fmaxf(a1.z, a1.w)));
    #pragma unroll
    for (int off = 16; off >= 1; off >>= 1) {
        mn = fminf(mn, __shfl_xor(mn, off, 32));
        mx = fmaxf(mx, __shfl_xor(mx, off, 32));
    }
    const float inv = 1.5707963267948966f / (mx - mn + 1e-8f);   // * pi/2

    // ---- statevector |0..0>, 8 complex amps per lane ----
    float aR[8], aI[8];
    #pragma unroll
    for (int k = 0; k < 8; ++k) { aR[k] = 0.f; aI[k] = 0.f; }
    if (lane == 0) aR[0] = 1.0f;

    // ---- encoding layer: RY(angle_q) on qubit q; x[row][0..7] live in lane 0 ----
    float xq[8] = {a0.x, a0.y, a0.z, a0.w, a1.x, a1.y, a1.z, a1.w};
    #pragma unroll
    for (int q = 0; q < NQ; ++q) {
        float v   = __shfl(xq[q], 0, 32);
        float ang = (v - mn) * inv;
        float s, c;
        __sincosf(0.5f * ang, &s, &c);
        ry_gate(aR, aI, q, c, s, lane);
    }

    // ---- variational layers ----
    #pragma unroll
    for (int l = 0; l < NLAYERS; ++l) {
        #pragma unroll
        for (int q = 0; q < NQ; ++q) {
            float s, c;
            __sincosf(0.5f * theta[l * NQ + q], &s, &c);
            ry_gate(aR, aI, q, c, s, lane);
        }
        #pragma unroll
        for (int q = 0; q < NQ; ++q)
            cnot_gate(aR, aI, q, (q + 1) & 7, lane);
        #pragma unroll
        for (int q = 0; q < NQ; ++q) {
            float s, c;
            __sincosf(0.5f * phi[l * NQ + q], &s, &c);
            rz_gate(aR, aI, q, c, s, lane);
        }
    }

    // ---- Z expectations: z_q = sum_i (1 - 2*bit_q(i)) |amp_i|^2 ----
    float z[8];
    float ptot = 0.f;
    z[0] = z[1] = z[2] = 0.f;
    #pragma unroll
    for (int k = 0; k < 8; ++k) {
        float p = aR[k] * aR[k] + aI[k] * aI[k];
        ptot += p;
        z[0] += (k & 1) ? -p : p;
        z[1] += (k & 2) ? -p : p;
        z[2] += (k & 4) ? -p : p;
    }
    #pragma unroll
    for (int q = 3; q < 8; ++q) z[q] = ((lane >> (q - 3)) & 1) ? -ptot : ptot;
    #pragma unroll
    for (int off = 16; off >= 1; off >>= 1) {
        #pragma unroll
        for (int q = 0; q < 8; ++q) z[q] += __shfl_xor(z[q], off, 32);
    }
    if (lane == 0) {
        #pragma unroll
        for (int q = 0; q < 8; ++q) qLDS[wave * NQ + q] = z[q];
    }
    __syncthreads();

    // ---- h = relu(q @ W1 + b1), split into bf16 hi/lo in LDS (16 x 128) ----
    #pragma unroll
    for (int e = 0; e < 4; ++e) {
        const int idx = tid * 4 + e;        // 0..2047
        const int r   = idx >> 7;
        const int cc  = idx & (HDIM - 1);
        float acc = b1[cc];
        #pragma unroll
        for (int k = 0; k < NQ; ++k) acc += qLDS[r * NQ + k] * W1[k * HDIM + cc];
        acc = fmaxf(acc, 0.f);
        __bf16 hh = (__bf16)acc;
        hhi[idx] = hh;
        hlo[idx] = (__bf16)(acc - (float)hh);
    }
    __syncthreads();

    // ---- transformed = h @ W2 (M=16, N=256, K=128); wave w owns cols [16w,16w+16) ----
    // Split bf16 WMMA: acc += hi*hi + hi*lo + lo*hi  (~fp32 accuracy, fp32 accumulate)
    const int m    = lane & 15;
    const int half = lane >> 4;
    const int ncol = wave * 16 + m;

    v8f acc = {};
    #pragma unroll
    for (int k0 = 0; k0 < HDIM; k0 += 32) {
        v16bf ahi, alo, bhi, blo;
        #pragma unroll
        for (int v = 0; v < 8; ++v) {
            // ISA 7.12.2: 16-bit A 16x32 — VGPR v holds K = {2v, 2v+1} (+16 for v>=4), +8 for lanes 16-31
            const int kl = ((v < 4) ? (2 * v) : (16 + 2 * (v - 4))) + 8 * half;
            const int ka = k0 + kl;
            ahi[2 * v]     = hhi[m * HDIM + ka];
            ahi[2 * v + 1] = hhi[m * HDIM + ka + 1];
            alo[2 * v]     = hlo[m * HDIM + ka];
            alo[2 * v + 1] = hlo[m * HDIM + ka + 1];
            // B assumed symmetric to A: lane%16 = column N, same K packing (L2-resident W2)
            float w0 = W2[(size_t)ka * DDIM + ncol];
            float w1 = W2[(size_t)(ka + 1) * DDIM + ncol];
            __bf16 w0h = (__bf16)w0;
            __bf16 w1h = (__bf16)w1;
            bhi[2 * v]     = w0h;
            bhi[2 * v + 1] = w1h;
            blo[2 * v]     = (__bf16)(w0 - (float)w0h);
            blo[2 * v + 1] = (__bf16)(w1 - (float)w1h);
        }
        acc = __builtin_amdgcn_wmma_f32_16x16x32_bf16(false, ahi, false, bhi, (short)0, acc, false, false);
        acc = __builtin_amdgcn_wmma_f32_16x16x32_bf16(false, ahi, false, blo, (short)0, acc, false, false);
        acc = __builtin_amdgcn_wmma_f32_16x16x32_bf16(false, alo, false, bhi, (short)0, acc, false, false);
    }

    // ---- epilogue: out = x + transformed + b2 ; C/D layout: M = r + 8*half, N = lane%16 ----
    const float bb = b2[ncol];
    #pragma unroll
    for (int r8 = 0; r8 < 8; ++r8) {
        const long grow = (long)blockIdx.x * 16 + r8 + 8 * half;
        const size_t o  = (size_t)grow * DDIM + ncol;
        out[o] = x[o] + acc[r8] + bb;
    }
}

extern "C" void kernel_launch(void* const* d_in, const int* in_sizes, int n_in,
                              void* d_out, int out_size, void* d_ws, size_t ws_size,
                              hipStream_t stream) {
    const float* x     = (const float*)d_in[0];
    const float* theta = (const float*)d_in[1];
    const float* phi   = (const float*)d_in[2];
    const float* W1    = (const float*)d_in[3];
    const float* b1    = (const float*)d_in[4];
    const float* W2    = (const float*)d_in[5];
    const float* b2    = (const float*)d_in[6];
    float* out = (float*)d_out;

    const int B = in_sizes[0] / DDIM;       // 16384
    dim3 grid(B / 16);
    vqc_fused_kernel<<<grid, 512, 0, stream>>>(x, theta, phi, W1, b1, W2, b2, out);
}
